// TokenMixer_85959475462854
// MI455X (gfx1250) — matrix operand order repaired and checked by reading, not compiled, not run on hardware
//
#include <hip/hip_runtime.h>
#include <hip/hip_bf16.h>
#include <math.h>

// MI455X / gfx1250, wave32. Core GEMM: v_wmma_f32_16x16x32_bf16 with a
// double-buffered async (GLOBAL_LOAD_ASYNC_TO_LDS_B128 / ASYNCcnt) pipeline.
// All matmuls canonicalized to: C[M,N] = epi( A[M,K](row) x Bt[N,K](row)^T + bias )

typedef __bf16 bf16_t;
typedef __attribute__((ext_vector_type(16))) __bf16 v16bf;
typedef __attribute__((ext_vector_type(8)))  __bf16 v8bf;
typedef __attribute__((ext_vector_type(8)))  float  v8f;
typedef int v4i_t __attribute__((vector_size(16)));   // matches builtin param type

#define LDS_STRIDE 40   // 32 K-elems + 8 pad (bf16); row pitch 80B keeps 16B alignment

#if __has_builtin(__builtin_amdgcn_global_load_async_to_lds_b128) && \
    __has_builtin(__builtin_amdgcn_s_wait_asynccnt)
#define USE_ASYNC_LDS 1
#else
#define USE_ASYNC_LDS 0
#endif

#define AS_GLOBAL __attribute__((address_space(1)))
#define AS_LDS    __attribute__((address_space(3)))

union Frag16 { v16bf v; v8bf h[2]; };
union AccU   { v8f   v; float f[8]; };

__device__ __forceinline__ float gelu_exact(float x) {
    // erf-based GELU, matches jax.nn.gelu(approximate=False)
    return 0.5f * x * (1.0f + erff(x * 0.70710678118654752f));
}

// ---------------------------------------------------------------- converts
__global__ void cvt_f32_bf16(const float* __restrict__ in, bf16_t* __restrict__ out, int n) {
    int i = blockIdx.x * blockDim.x + threadIdx.x;
    int stride = gridDim.x * blockDim.x;
    for (; i < n; i += stride) out[i] = (bf16_t)in[i];
}

// out[b][c][r] = (bf16) in[b][r][c]   (rows x cols -> cols x rows), 32x32 LDS tiles
__global__ void transpose_cvt(const float* __restrict__ in, bf16_t* __restrict__ out,
                              int rows, int cols) {
    __shared__ float tile[32][33];
    const float* inb  = in  + (size_t)blockIdx.z * rows * cols;
    bf16_t*      outb = out + (size_t)blockIdx.z * rows * cols;
    int c  = blockIdx.x * 32 + threadIdx.x;
    int r0 = blockIdx.y * 32;
    #pragma unroll
    for (int j = 0; j < 32; j += 8)
        tile[threadIdx.y + j][threadIdx.x] = inb[(size_t)(r0 + threadIdx.y + j) * cols + c];
    __syncthreads();
    int rOut = blockIdx.x * 32 + threadIdx.y;
    int cOut = r0 + threadIdx.x;
    #pragma unroll
    for (int j = 0; j < 32; j += 8)
        outb[(size_t)(rOut + j) * rows + cOut] = (bf16_t)tile[threadIdx.x][threadIdx.y + j];
}

// ---------------------------------------------------------------- WMMA GEMM
// OUT_MODE: 0 = bf16 C[m*ldc+n], 1 = bf16 transposed C[n*ldc+m], 2 = f32 C[m*ldc+n]
template<int OUT_MODE, bool GELU_EPI, bool HAS_BIAS>
__global__ __launch_bounds__(256)
void gemm_bf16_wmma(const bf16_t* __restrict__ A,  long long strideA, int lda,
                    const bf16_t* __restrict__ Bt, long long strideB, int ldb,
                    const float*  __restrict__ bias,
                    void* __restrict__ C, long long strideC, int ldc, int K)
{
    const int tid  = threadIdx.x;
    const int lane = tid & 31;
    const int wave = tid >> 5;      // 8 waves
    const int wm   = wave >> 1;     // 0..3 : M direction (32 rows each)
    const int wn   = wave & 1;      // 0..1 : N direction (64 cols each)
    const int h    = lane >> 4;     // lane half
    const int lr   = lane & 15;

    const bf16_t* Ab = A  + (long long)blockIdx.z * strideA + (long long)blockIdx.y * 128 * lda;
    const bf16_t* Bb = Bt + (long long)blockIdx.z * strideB + (long long)blockIdx.x * 128 * ldb;

    v8f acc[2][4] = {};
    const int nk = K >> 5;          // number of K=32 chunks

    // per-thread tile-fill assignment: 128 rows x 4 x 16B chunks, 2 chunks each
    const int rowc0 = tid >> 2,           segc0 = tid & 3;
    const int rowc1 = (tid + 256) >> 2,   segc1 = tid & 3;

#if USE_ASYNC_LDS
    // ---------------- double-buffered async global->LDS pipeline ----------------
    __shared__ __align__(16) bf16_t sA[2][128 * LDS_STRIDE];
    __shared__ __align__(16) bf16_t sB[2][128 * LDS_STRIDE];

    auto issue_fill = [&](int kk, int buf) {
        // 4 async ops/thread/chunk (ASYNCcnt +4), each copies 16B global->LDS
        __builtin_amdgcn_global_load_async_to_lds_b128(
            (AS_GLOBAL v4i_t*)(Ab + (long long)rowc0 * lda + kk + segc0 * 8),
            (AS_LDS   v4i_t*)(&sA[buf][rowc0 * LDS_STRIDE + segc0 * 8]), 0, 0);
        __builtin_amdgcn_global_load_async_to_lds_b128(
            (AS_GLOBAL v4i_t*)(Bb + (long long)rowc0 * ldb + kk + segc0 * 8),
            (AS_LDS   v4i_t*)(&sB[buf][rowc0 * LDS_STRIDE + segc0 * 8]), 0, 0);
        __builtin_amdgcn_global_load_async_to_lds_b128(
            (AS_GLOBAL v4i_t*)(Ab + (long long)rowc1 * lda + kk + segc1 * 8),
            (AS_LDS   v4i_t*)(&sA[buf][rowc1 * LDS_STRIDE + segc1 * 8]), 0, 0);
        __builtin_amdgcn_global_load_async_to_lds_b128(
            (AS_GLOBAL v4i_t*)(Bb + (long long)rowc1 * ldb + kk + segc1 * 8),
            (AS_LDS   v4i_t*)(&sB[buf][rowc1 * LDS_STRIDE + segc1 * 8]), 0, 0);
    };

    auto compute_chunk = [&](int buf) {
        Frag16 fa[2];
        #pragma unroll
        for (int im = 0; im < 2; ++im) {
            int row = wm * 32 + im * 16 + lr;
            fa[im].h[0] = *(const v8bf*)(&sA[buf][row * LDS_STRIDE + 8 * h]);
            fa[im].h[1] = *(const v8bf*)(&sA[buf][row * LDS_STRIDE + 16 + 8 * h]);
        }
        Frag16 fb[4];
        #pragma unroll
        for (int jn = 0; jn < 4; ++jn) {
            int row = wn * 64 + jn * 16 + lr;
            fb[jn].h[0] = *(const v8bf*)(&sB[buf][row * LDS_STRIDE + 8 * h]);
            fb[jn].h[1] = *(const v8bf*)(&sB[buf][row * LDS_STRIDE + 16 + 8 * h]);
        }
        #pragma unroll
        for (int im = 0; im < 2; ++im)
            #pragma unroll
            for (int jn = 0; jn < 4; ++jn)
                acc[im][jn] = __builtin_amdgcn_wmma_f32_16x16x32_bf16(
                    false, fa[im].v, false, fb[jn].v,
                    (short)0, acc[im][jn], false, false);
    };

    issue_fill(0, 0);
    if (nk > 1) issue_fill(32, 1);

    // steady state: branch-free waits, 2 chunks always in flight
    int ik = 0;
    for (; ik + 2 < nk; ++ik) {
        __builtin_amdgcn_s_wait_asynccnt(4);   // retire chunk ik's 4 ops (in-order)
        __syncthreads();
        compute_chunk(ik & 1);
        __syncthreads();                       // everyone done reading this buffer
        issue_fill((ik + 2) * 32, ik & 1);     // overlaps next chunk's compute
    }
    // tail: at most 2 chunks remain in flight, nothing more to issue
    if (ik < nk && ik + 1 < nk) {
        __builtin_amdgcn_s_wait_asynccnt(4);
        __syncthreads();
        compute_chunk(ik & 1);
        __syncthreads();
        ++ik;
    }
    if (ik < nk) {
        __builtin_amdgcn_s_wait_asynccnt(0);
        __syncthreads();
        compute_chunk(ik & 1);
    }
#else
    // ---------------- register-staged software pipeline (fallback) ----------------
    __shared__ __align__(16) bf16_t sA[128 * LDS_STRIDE];
    __shared__ __align__(16) bf16_t sB[128 * LDS_STRIDE];

    float4 ra0, ra1, rb0, rb1;
    auto gload = [&](int kk) {
        ra0 = *(const float4*)(Ab + (long long)rowc0 * lda + kk + segc0 * 8);
        rb0 = *(const float4*)(Bb + (long long)rowc0 * ldb + kk + segc0 * 8);
        ra1 = *(const float4*)(Ab + (long long)rowc1 * lda + kk + segc1 * 8);
        rb1 = *(const float4*)(Bb + (long long)rowc1 * ldb + kk + segc1 * 8);
    };
    gload(0);
    for (int ik = 0; ik < nk; ++ik) {
        *(float4*)(&sA[rowc0 * LDS_STRIDE + segc0 * 8]) = ra0;
        *(float4*)(&sB[rowc0 * LDS_STRIDE + segc0 * 8]) = rb0;
        *(float4*)(&sA[rowc1 * LDS_STRIDE + segc1 * 8]) = ra1;
        *(float4*)(&sB[rowc1 * LDS_STRIDE + segc1 * 8]) = rb1;
        __syncthreads();
        if (ik + 1 < nk) gload((ik + 1) * 32);   // in flight during WMMAs

        Frag16 fa[2];
        #pragma unroll
        for (int im = 0; im < 2; ++im) {
            int row = wm * 32 + im * 16 + lr;
            fa[im].h[0] = *(const v8bf*)(&sA[row * LDS_STRIDE + 8 * h]);
            fa[im].h[1] = *(const v8bf*)(&sA[row * LDS_STRIDE + 16 + 8 * h]);
        }
        Frag16 fb[4];
        #pragma unroll
        for (int jn = 0; jn < 4; ++jn) {
            int row = wn * 64 + jn * 16 + lr;
            fb[jn].h[0] = *(const v8bf*)(&sB[row * LDS_STRIDE + 8 * h]);
            fb[jn].h[1] = *(const v8bf*)(&sB[row * LDS_STRIDE + 16 + 8 * h]);
        }
        #pragma unroll
        for (int im = 0; im < 2; ++im)
            #pragma unroll
            for (int jn = 0; jn < 4; ++jn)
                acc[im][jn] = __builtin_amdgcn_wmma_f32_16x16x32_bf16(
                    false, fa[im].v, false, fb[jn].v,
                    (short)0, acc[im][jn], false, false);
        __syncthreads();
    }
#endif

    // epilogue: C/D layout -> lane has N=lr, VGPR r has M = r + 8h (per 16x16 tile)
    #pragma unroll
    for (int im = 0; im < 2; ++im) {
        #pragma unroll
        for (int jn = 0; jn < 4; ++jn) {
            AccU a; a.v = acc[im][jn];
            int m_base = blockIdx.y * 128 + wm * 32 + im * 16 + 8 * h;
            int n_g    = blockIdx.x * 128 + wn * 64 + jn * 16 + lr;
            float bval = HAS_BIAS ? bias[n_g] : 0.0f;
            if (OUT_MODE == 1) {
                // transposed bf16: per lane M is contiguous -> one 16B store
                bf16_t* Cb = (bf16_t*)C + blockIdx.z * strideC;
                v8bf pk;
                #pragma unroll
                for (int r = 0; r < 8; ++r) {
                    float v = a.f[r] + bval;
                    if (GELU_EPI) v = gelu_exact(v);
                    pk[r] = (bf16_t)v;
                }
                *(v8bf*)(&Cb[(long long)n_g * ldc + m_base]) = pk;
            } else {
                #pragma unroll
                for (int r = 0; r < 8; ++r) {
                    float v = a.f[r] + bval;
                    if (GELU_EPI) v = gelu_exact(v);
                    long long m_g = m_base + r;
                    if (OUT_MODE == 2) {
                        float* Cb = (float*)C + blockIdx.z * strideC;
                        Cb[m_g * ldc + n_g] = v;
                    } else {
                        bf16_t* Cb = (bf16_t*)C + blockIdx.z * strideC;
                        Cb[m_g * ldc + n_g] = (bf16_t)v;
                    }
                }
            }
        }
    }
}

// ---------------------------------------------------------------- launcher
extern "C" void kernel_launch(void* const* d_in, const int* in_sizes, int n_in,
                              void* d_out, int out_size, void* d_ws, size_t ws_size,
                              hipStream_t stream)
{
    constexpr int BATCH = 8, NTOK = 4096, D = 512, DP = 512;
    constexpr long long MT = (long long)BATCH * NTOK;  // 32768 tokens flattened

    const float* X   = (const float*)d_in[0];
    const float* W1a = (const float*)d_in[1];
    const float* b1a = (const float*)d_in[2];
    const float* W1b = (const float*)d_in[3];
    const float* b1b = (const float*)d_in[4];
    const float* W2a = (const float*)d_in[5];
    const float* b2a = (const float*)d_in[6];
    const float* W2b = (const float*)d_in[7];
    const float* b2b = (const float*)d_in[8];
    float* Y = (float*)d_out;

    char* ws = (char*)d_ws;
    size_t off = 0;
    auto alloc = [&](size_t bytes) -> void* {
        void* p = ws + off; off += (bytes + 255) & ~(size_t)255; return p;
    };
    bf16_t* Xbf  = (bf16_t*)alloc((size_t)MT * D * 2);        // X, bf16, [MT][D]
    bf16_t* XT   = (bf16_t*)alloc((size_t)MT * D * 2);        // X^T per batch, [b][D][NTOK]
    bf16_t* W1ab = (bf16_t*)alloc((size_t)D * D * 2);         // weights keep (out,in)=Bt layout
    bf16_t* W1bb = (bf16_t*)alloc((size_t)DP * D * 2);
    bf16_t* W2ab = (bf16_t*)alloc((size_t)D * D * 2);
    bf16_t* W2bb = (bf16_t*)alloc((size_t)DP * D * 2);
    bf16_t* Hbuf = (bf16_t*)alloc((size_t)MT * D * 2);        // GELU(X@Wa^T+ba), reused
    bf16_t* W1T  = (bf16_t*)alloc((size_t)DP * MT * 2);       // W1 transposed [DP][MT]
    bf16_t* W2m  = (bf16_t*)alloc((size_t)MT * DP * 2);       // W2 row-major [MT][DP]
    bf16_t* ATb  = (bf16_t*)alloc((size_t)BATCH * D * DP * 2);// GELU(P)^T per batch [b][D][DP]

    // precision converts + X transpose
    cvt_f32_bf16<<<1024, 256, 0, stream>>>(X,   Xbf,  (int)(MT * D));
    cvt_f32_bf16<<<256, 256, 0, stream>>>(W1a, W1ab, D * D);
    cvt_f32_bf16<<<256, 256, 0, stream>>>(W1b, W1bb, DP * D);
    cvt_f32_bf16<<<256, 256, 0, stream>>>(W2a, W2ab, D * D);
    cvt_f32_bf16<<<256, 256, 0, stream>>>(W2b, W2bb, DP * D);
    transpose_cvt<<<dim3(D / 32, NTOK / 32, BATCH), dim3(32, 8), 0, stream>>>(X, XT, NTOK, D);

    dim3 blk(256);
    // MLP1: H = GELU(X @ W1a^T + b1a)
    gemm_bf16_wmma<0, true, true><<<dim3(D / 128, MT / 128, 1), blk, 0, stream>>>(
        Xbf, 0, D, W1ab, 0, D, b1a, Hbuf, 0, D, D);
    // W1 = H @ W1b^T + b1b, stored transposed: W1T[p][token]
    gemm_bf16_wmma<1, false, true><<<dim3(DP / 128, MT / 128, 1), blk, 0, stream>>>(
        Hbuf, 0, D, W1bb, 0, D, b1b, W1T, 0, (int)MT, D);
    // MLP2: H = GELU(X @ W2a^T + b2a)
    gemm_bf16_wmma<0, true, true><<<dim3(D / 128, MT / 128, 1), blk, 0, stream>>>(
        Xbf, 0, D, W2ab, 0, D, b2a, Hbuf, 0, D, D);
    // W2 = H @ W2b^T + b2b, stored row-major [MT][DP]
    gemm_bf16_wmma<0, false, true><<<dim3(DP / 128, MT / 128, 1), blk, 0, stream>>>(
        Hbuf, 0, D, W2bb, 0, D, b2b, W2m, 0, DP, D);
    // P_b = W1T_b @ X_b  (K = NTOK), A' = GELU(P) stored transposed -> ATb[b][d][p]
    gemm_bf16_wmma<1, true, false><<<dim3(D / 128, DP / 128, BATCH), blk, 0, stream>>>(
        W1T, (long long)NTOK, (int)MT,          // A = W1T + b*4096, lda = 32768
        XT, (long long)NTOK * D, NTOK,          // Bt = X^T per batch, ldb = 4096
        nullptr, ATb, (long long)D * DP, DP, NTOK);
    // Y_b = W2_b @ A_b  (Bt = A^T), f32 out
    gemm_bf16_wmma<2, false, false><<<dim3(D / 128, NTOK / 128, BATCH), blk, 0, stream>>>(
        W2m, (long long)NTOK * DP, DP,
        ATb, (long long)D * DP, DP,
        nullptr, Y, (long long)NTOK * D, D, DP);
    (void)in_sizes; (void)n_in; (void)out_size; (void)ws_size;
}
